// AggregatorSubLayer_34488587387574
// MI455X (gfx1250) — compile-verified, matrix operand order checked by reading, not run on hardware
//
#include <hip/hip_runtime.h>

// relu(concat(review, user[uidx], item[iidx]) @ W)  -> [N, 128] f32
// f16 WMMA (16x16x32), f32 accumulation. W (384x128 f32) is converted to f16,
// transposed, and kept fully resident in LDS (12 chunks x 128x40 f16 = 120 KB,
// within the 320 KB WGP budget). Persistent blocks grid-stride over 256-row
// blocks; the hot loop is pure global A-load/convert + ds_load_b128 B + WMMA.

#define D_EMB   128
#define KCHUNK  32
#define KPAD    40   // 32 + 8 f16 pad: 80B row stride, 16B aligned, conflict-free
#define NCHUNK  12   // 384 / 32

typedef _Float16 v16h __attribute__((ext_vector_type(16)));
typedef float    v8f  __attribute__((ext_vector_type(8)));
typedef _Float16 h2   __attribute__((ext_vector_type(2)));

union AFrag { v16h v; h2 h[8]; };
union BFrag { v16h v; float4 f[2]; };

// Per-lane A fragment for v_wmma_f32_16x16x32_f16:
// lanes 0-15 (M=lane):    K = {0..7} in VGPR0-3, {16..23} in VGPR4-7
// lanes 16-31 (M=lane-16): K = {8..15},          {24..31}
// -> two contiguous 8xf32 runs per lane, packed to f16 pairs.
__device__ __forceinline__ v16h load_a_frag(const float* __restrict__ p) {
  float4 f0 = *(const float4*)(p + 0);
  float4 f1 = *(const float4*)(p + 4);
  float4 f2 = *(const float4*)(p + 16);
  float4 f3 = *(const float4*)(p + 20);
  AFrag u;
  u.h[0] = (h2)__builtin_amdgcn_cvt_pkrtz(f0.x, f0.y);
  u.h[1] = (h2)__builtin_amdgcn_cvt_pkrtz(f0.z, f0.w);
  u.h[2] = (h2)__builtin_amdgcn_cvt_pkrtz(f1.x, f1.y);
  u.h[3] = (h2)__builtin_amdgcn_cvt_pkrtz(f1.z, f1.w);
  u.h[4] = (h2)__builtin_amdgcn_cvt_pkrtz(f2.x, f2.y);
  u.h[5] = (h2)__builtin_amdgcn_cvt_pkrtz(f2.z, f2.w);
  u.h[6] = (h2)__builtin_amdgcn_cvt_pkrtz(f3.x, f3.y);
  u.h[7] = (h2)__builtin_amdgcn_cvt_pkrtz(f3.z, f3.w);
  return u.v;
}

// B fragment (K x N = 32 x 16): lane = column N (mod 16), per-lane contiguous
// K runs of 8 (same split as A) -> two ds_load_b128 from the transposed tile.
__device__ __forceinline__ v16h load_b_frag(const _Float16* q) {
  BFrag u;
  u.f[0] = *(const float4*)(q);
  u.f[1] = *(const float4*)(q + 16);
  return u.v;
}

// One thread converts column n, k-local range [half*16, half*16+16) of chunk c.
// Global reads: for fixed j, lanes cover consecutive n -> contiguous 512B.
// LDS writes: 16 f16 contiguous -> two ds_store_b128. Fixed trip counts, no guards.
__device__ __forceinline__ void convert_w_chunk(const float* __restrict__ Wg, int c,
                                                _Float16* dst, int n, int half) {
  const float* src = Wg + ((size_t)c * KCHUNK + half * 16) * D_EMB + n;
  float f[16];
  #pragma unroll
  for (int j = 0; j < 16; ++j) f[j] = src[(size_t)j * D_EMB];
  union { h2 h[8]; float4 q[2]; } u;
  #pragma unroll
  for (int j = 0; j < 8; ++j)
    u.h[j] = (h2)__builtin_amdgcn_cvt_pkrtz(f[2 * j], f[2 * j + 1]);
  _Float16* d = dst + n * KPAD + half * 16;
  *(float4*)(d)     = u.q[0];
  *(float4*)(d + 8) = u.q[1];
}

__global__ __launch_bounds__(256) void agg_wmma_kernel(
    const float* __restrict__ review,   // [N, 128]
    const float* __restrict__ item,     // [N_items, 128]
    const float* __restrict__ user,     // [N_users, 128]
    const int*   __restrict__ uidx,     // [N]
    const int*   __restrict__ iidx,     // [N]
    const float* __restrict__ Wg,       // [384, 128]
    float*       __restrict__ out,      // [N, 128]
    int n_reviews) {
  __shared__ _Float16 sW[NCHUNK][D_EMB * KPAD];   // 120 KB, resident all kernel

  const int tid = threadIdx.x;
  {
    const int n = tid & (D_EMB - 1);
    const int half = tid >> 7;
    #pragma unroll
    for (int c = 0; c < NCHUNK; ++c) convert_w_chunk(Wg, c, sW[c], n, half);
  }
  __syncthreads();

  const int lane = tid & 31;
  const int wave = tid >> 5;
  const int m    = lane & 15;
  const int hi   = lane >> 4;
  const int koff = hi * 8;
  const int num_rb = (n_reviews + 255) >> 8;
  const int rmax = n_reviews - 1;

  for (int rb = blockIdx.x; rb < num_rb; rb += gridDim.x) {
    const int rows_base = rb * 256 + wave * 32;

    int r0 = rows_base + m;       if (r0 > rmax) r0 = rmax;
    int r1 = rows_base + 16 + m;  if (r1 > rmax) r1 = rmax;

    const float* pr0 = review + (size_t)r0 * D_EMB;
    const float* pr1 = review + (size_t)r1 * D_EMB;
    const float* pu0 = user   + (size_t)uidx[r0] * D_EMB;
    const float* pu1 = user   + (size_t)uidx[r1] * D_EMB;
    const float* pi0 = item   + (size_t)iidx[r0] * D_EMB;
    const float* pi1 = item   + (size_t)iidx[r1] * D_EMB;

    v8f acc0[8], acc1[8];
    #pragma unroll
    for (int nt = 0; nt < 8; ++nt) {
      acc0[nt] = (v8f){0.f, 0.f, 0.f, 0.f, 0.f, 0.f, 0.f, 0.f};
      acc1[nt] = (v8f){0.f, 0.f, 0.f, 0.f, 0.f, 0.f, 0.f, 0.f};
    }

    #pragma unroll
    for (int c = 0; c < NCHUNK; ++c) {
      const int seg = c >> 2;      // 0: review, 1: user, 2: item
      const int cc  = c & 3;       // 32-K chunk within the 128-wide segment
      const float* q0 = (seg == 0) ? pr0 : (seg == 1) ? pu0 : pi0;
      const float* q1 = (seg == 0) ? pr1 : (seg == 1) ? pu1 : pi1;

      v16h a0 = load_a_frag(q0 + cc * KCHUNK + koff);
      v16h a1 = load_a_frag(q1 + cc * KCHUNK + koff);

      const _Float16* buf = sW[c];
      #pragma unroll
      for (int nt = 0; nt < 8; ++nt) {
        v16h b = load_b_frag(buf + (nt * 16 + m) * KPAD + koff);
        acc0[nt] = __builtin_amdgcn_wmma_f32_16x16x32_f16(
            false, a0, false, b, (short)0, acc0[nt], false, false);
        acc1[nt] = __builtin_amdgcn_wmma_f32_16x16x32_f16(
            false, a1, false, b, (short)0, acc1[nt], false, false);
      }
    }

    // C/D layout: VGPR v -> row (hi*8 + v), lane -> column m. ReLU + store.
    #pragma unroll
    for (int mt = 0; mt < 2; ++mt) {
      const int row_b = rows_base + mt * 16 + hi * 8;
      const v8f* acc = mt ? acc1 : acc0;
      #pragma unroll
      for (int v = 0; v < 8; ++v) {
        const int r = row_b + v;
        if (r < n_reviews) {
          float* orow = out + (size_t)r * D_EMB + m;
          #pragma unroll
          for (int nt = 0; nt < 8; ++nt) {
            float x = acc[nt][v];
            orow[nt * 16] = x > 0.f ? x : 0.f;
          }
        }
      }
    }
  }
}

extern "C" void kernel_launch(void* const* d_in, const int* in_sizes, int n_in,
                              void* d_out, int out_size, void* d_ws, size_t ws_size,
                              hipStream_t stream) {
  (void)n_in; (void)out_size; (void)d_ws; (void)ws_size;
  const float* review = (const float*)d_in[0];
  const float* item   = (const float*)d_in[1];
  const float* user   = (const float*)d_in[2];
  const int*   uidx   = (const int*)d_in[3];
  const int*   iidx   = (const int*)d_in[4];
  const float* Wg     = (const float*)d_in[5];
  float* out = (float*)d_out;

  const int n_reviews = in_sizes[3];        // len(adj_user_idx)
  const int num_rb = (n_reviews + 255) / 256;
  const int blocks = num_rb < 1024 ? num_rb : 1024;
  agg_wmma_kernel<<<blocks, 256, 0, stream>>>(review, item, user, uidx, iidx,
                                              Wg, out, n_reviews);
}
